// GPTNeoX_12463995093281
// MI455X (gfx1250) — matrix operand order repaired and checked by reading, not compiled, not run on hardware
//
#include <hip/hip_runtime.h>

// ---------------- problem constants (from reference) ----------------
#define S_LEN   1024
#define HD      2048
#define NLAYER  2
#define NGROUP  4
#define GSIZE   8
#define HEADD   64
#define GD      (NGROUP * HEADD)     // 256
#define FFD     (4 * HD)             // 8192
#define VOCAB   38400

// ---------------- WMMA types (CDNA5 / gfx1250, wave32) ----------------
typedef __attribute__((ext_vector_type(16))) __bf16 v16bf;
typedef __attribute__((ext_vector_type(8)))  float  v8f;

union FragBF { v16bf v; unsigned int u[8]; };

__device__ __forceinline__ unsigned short f2bf(float f) {
  // round-to-nearest-even f32 -> bf16
  unsigned int u = __float_as_uint(f);
  u += 0x7FFFu + ((u >> 16) & 1u);
  return (unsigned short)(u >> 16);
}

__device__ __forceinline__ unsigned int pack2bf(float lo, float hi) {
  return (unsigned int)f2bf(lo) | ((unsigned int)f2bf(hi) << 16);
}

// =====================================================================
// bf16-WMMA GEMM:  Y[M,N] = act( scale * (A[M,K] @ B[K,N]) + bias )
//   REQUIRES: M % 128 == 0, N % 64 == 0, K % 64 == 0 (true for all calls)
//   A: bf16 row-major (lda in elements)  -> staged with ASYNC global->LDS
//   B: f32; btrans==0 -> K x N row-major (ldb); btrans==1 -> N x K (ldb)
//   Grid: blockIdx.x = M/128 (FASTEST -> the 8 blocks sharing one B
//         column-tile are adjacent in dispatch, so B streams from HBM ~once
//         and is served from L2 (192 MB) for the rest), blockIdx.y = N/64.
//   Workgroup: 256 threads (8 waves); tile 128(M) x 64(N); K step 64;
//   each wave: 2x2 accs, 8 v_wmma_f32_16x16x32_bf16 per staging round.
// =====================================================================
__global__ __launch_bounds__(256)
void wmma_gemm_kernel(const unsigned short* __restrict__ Abf, int lda,
                      const float* __restrict__ Bm, int ldb, int btrans,
                      const float* __restrict__ bias,
                      void* __restrict__ Yp, int ldc,
                      int M, int N, int K, float scale, int relu, int out_bf16)
{
  __shared__ __align__(16) unsigned short As[128 * 64];  // [m][k], k-major
  __shared__ __align__(16) unsigned short Bs[64 * 64];   // [n][k], k-major

  const int tid  = threadIdx.x;
  const int lane = tid & 31;
  const int wave = tid >> 5;
  const int wm   = wave >> 1;      // 0..3 : 32-row block within 128
  const int wn   = wave & 1;       // 0..1 : 32-col block within 64
  const int Mbase = blockIdx.x * 128;   // M fastest (B-column reuse in L2)
  const int Nbase = blockIdx.y * 64;
  const int half = lane >> 4;      // lane group (ISA A/B fragment layout)
  const int l16  = lane & 15;

  v8f acc[2][2];
#pragma unroll
  for (int a = 0; a < 2; ++a)
#pragma unroll
    for (int b = 0; b < 2; ++b)
#pragma unroll
      for (int r = 0; r < 8; ++r) acc[a][b][r] = 0.0f;

  for (int kk = 0; kk < K; kk += 64) {
    // ---- prefetch next A K-tile (global_prefetch_b8) ----
    if (kk + 64 < K) {
      int r = tid >> 1, c = (tid & 1) * 32;
      __builtin_prefetch(&Abf[(size_t)(Mbase + r) * lda + kk + 64 + c], 0, 1);
    }

    // ---- A tile 128x64 bf16: ASYNC global->LDS copy (no conversion) ----
    // 16 KB tile, 4x b128 per thread; tracked by ASYNCcnt.
#pragma unroll
    for (int i = 0; i < 4; ++i) {
      int slot = tid + i * 256;            // 1024 b128 slots
      int r    = slot >> 3;                // 8 x b128 per 64-elem row
      int c8   = (slot & 7) * 8;
      const unsigned short* gp = Abf + (size_t)(Mbase + r) * lda + kk + c8;
      unsigned int lp = (unsigned int)(size_t)&As[r * 64 + c8];
      asm volatile("global_load_async_to_lds_b128 %0, %1, off"
                   :: "v"(lp), "v"(gp) : "memory");
    }

    // ---- stage B tile 64x64 (f32 -> bf16) into k-major [n][k] ----
    if (btrans) {
      // source N x K row-major: contiguous in k -> float4 + b64 stores
#pragma unroll
      for (int i = 0; i < 4; ++i) {
        int slot = tid + i * 256;          // 1024 float4 slots
        int n    = slot >> 4;
        int c4   = (slot & 15) * 4;
        const float4 x = *(const float4*)(Bm + (size_t)(Nbase + n) * ldb + kk + c4);
        uint2 p; p.x = pack2bf(x.x, x.y); p.y = pack2bf(x.z, x.w);
        *(uint2*)&Bs[n * 64 + c4] = p;
      }
    } else {
      // source K x N row-major: rows k,k+1 at same n (64-wide coalesced),
      // pack the k-pair into one b32 store
#pragma unroll
      for (int i = 0; i < 8; ++i) {
        int slot = tid + i * 256;          // 2048 packed dwords
        int k    = (slot >> 6) * 2;        // 0,2,..,62
        int n    = slot & 63;
        float lo = Bm[(size_t)(kk + k) * ldb + Nbase + n];
        float hi = Bm[(size_t)(kk + k + 1) * ldb + Nbase + n];
        *(unsigned int*)&Bs[n * 64 + k] = pack2bf(lo, hi);
      }
    }

    // fence our async copies, then workgroup barrier
    asm volatile("s_wait_asynccnt 0x0" ::: "memory");
    __syncthreads();

    // ---- 2 sub-steps of K=32: build fragments, 4 WMMAs each ----
    const unsigned int* As32 = (const unsigned int*)As;
    const unsigned int* Bs32 = (const unsigned int*)Bs;
#pragma unroll
    for (int ks = 0; ks < 2; ++ks) {
      const int kb = ks * 32;
      FragBF af[2], bfg[2];
#pragma unroll
      for (int sm = 0; sm < 2; ++sm) {
        int rowA = wm * 32 + sm * 16 + l16;            // M = lane&15
#pragma unroll
        for (int i = 0; i < 8; ++i) {
          int k = kb + ((i & 4) ? 16 : 0) + 2 * (i & 3) + half * 8;
          af[sm].u[i] = As32[rowA * 32 + (k >> 1)];    // packed bf16 pair
        }
      }
#pragma unroll
      for (int sn = 0; sn < 2; ++sn) {
        int rowB = wn * 32 + sn * 16 + l16;            // N = lane&15
#pragma unroll
        for (int j = 0; j < 8; ++j) {
          int k = kb + ((j & 4) ? 16 : 0) + 2 * (j & 3) + half * 8;
          bfg[sn].u[j] = Bs32[rowB * 32 + (k >> 1)];
        }
      }
#pragma unroll
      for (int sm = 0; sm < 2; ++sm)
#pragma unroll
        for (int sn = 0; sn < 2; ++sn)
          acc[sm][sn] = __builtin_amdgcn_wmma_f32_16x16x32_bf16(
              false, af[sm].v, false, bfg[sn].v,
              (short)0, acc[sm][sn], false, false);
    }
    __syncthreads();
  }

  // ---- epilogue: C/D layout VGPR r -> M = r + 8*half, N = lane&15 ----
  float*          Yf = (float*)Yp;
  unsigned short* Yh = (unsigned short*)Yp;
#pragma unroll
  for (int sm = 0; sm < 2; ++sm)
#pragma unroll
    for (int sn = 0; sn < 2; ++sn) {
      int col = Nbase + wn * 32 + sn * 16 + l16;
#pragma unroll
      for (int r = 0; r < 8; ++r) {
        int row = Mbase + wm * 32 + sm * 16 + r + 8 * half;
        float v = acc[sm][sn][r] * scale;
        if (bias) v += bias[col];
        if (relu) v = fmaxf(v, 0.0f);
        if (out_bf16) Yh[(size_t)row * ldc + col] = f2bf(v);
        else          Yf[(size_t)row * ldc + col] = v;
      }
    }
}

// =====================================================================
// Embedding gather + pos add; emits f32 h (for residual) and bf16 hbf (GEMM A)
// =====================================================================
__global__ void embed_kernel(const int* __restrict__ ids,
                             const float* __restrict__ tok,
                             const float* __restrict__ pos,
                             float* __restrict__ h,
                             unsigned short* __restrict__ hbf)
{
  int s = blockIdx.x;
  int d = blockIdx.y * 256 + threadIdx.x;
  int id = ids[s];
  float v = tok[(size_t)id * HD + d] + pos[(size_t)s * HD + d];
  h[(size_t)s * HD + d]   = v;
  hbf[(size_t)s * HD + d] = f2bf(v);
}

// qsum[s, g*64+d] = sum_j q[s, g*512 + j*64 + d]  (pre-reduce GQA subheads)
__global__ void qreduce_kernel(const float* __restrict__ q,
                               unsigned short* __restrict__ qsbf)
{
  int s = blockIdx.x;
  int t = threadIdx.x;             // 0..255 == g*64+d
  int g = t >> 6, d = t & 63;
  const float* qp = q + (size_t)s * HD + g * (GSIZE * HEADD) + d;
  float a = 0.0f;
#pragma unroll
  for (int j = 0; j < GSIZE; ++j) a += qp[j * HEADD];
  qsbf[(size_t)s * GD + t] = f2bf(a);
}

// Wo_fold[g*64+d, n] = sum_j Wo[g*512 + j*64 + d, n]  (fold ctx broadcast)
__global__ void wofold_kernel(const float* __restrict__ Wo, float* __restrict__ Wf)
{
  int n = blockIdx.x * 256 + threadIdx.x;   // 0..2047
  int r = blockIdx.y;                        // 0..255
  int g = r >> 6, d = r & 63;
  float s = 0.0f;
#pragma unroll
  for (int j = 0; j < GSIZE; ++j)
    s += Wo[(size_t)(g * (GSIZE * HEADD) + j * HEADD + d) * HD + n];
  Wf[(size_t)r * HD + n] = s;
}

// Row softmax over T=1024; writes normalized probs f32 in place + bf16 copy.
__global__ __launch_bounds__(256)
void softmax_kernel(float* __restrict__ scores, unsigned short* __restrict__ pbf)
{
  __shared__ float red[256];
  float* row          = scores + (size_t)blockIdx.x * S_LEN;
  unsigned short* rbf = pbf    + (size_t)blockIdx.x * S_LEN;
  int tid = threadIdx.x;
  float m = -3.4e38f;
  for (int i = tid; i < S_LEN; i += 256) m = fmaxf(m, row[i]);
  red[tid] = m; __syncthreads();
  for (int o = 128; o > 0; o >>= 1) { if (tid < o) red[tid] = fmaxf(red[tid], red[tid + o]); __syncthreads(); }
  m = red[0]; __syncthreads();
  float sum = 0.0f;
  for (int i = tid; i < S_LEN; i += 256) { float e = __expf(row[i] - m); row[i] = e; sum += e; }
  red[tid] = sum; __syncthreads();
  for (int o = 128; o > 0; o >>= 1) { if (tid < o) red[tid] += red[tid + o]; __syncthreads(); }
  float inv = 1.0f / red[0];
  for (int i = tid; i < S_LEN; i += 256) {
    float v = row[i] * inv;
    row[i] = v;
    rbf[i] = f2bf(v);
  }
}

// out = LayerNorm(x (+ res), gamma, beta); emits f32 out + bf16 outbf.
__global__ __launch_bounds__(256)
void add_ln_kernel(const float* __restrict__ x, const float* __restrict__ res,
                   const float* __restrict__ gamma, const float* __restrict__ beta,
                   float* __restrict__ out, unsigned short* __restrict__ outbf)
{
  __shared__ float red[256];
  const int tid = threadIdx.x;
  const size_t base = (size_t)blockIdx.x * HD;
  float v[8];
  float s = 0.0f;
#pragma unroll
  for (int i = 0; i < 8; ++i) {
    int c = tid + i * 256;
    float t = x[base + c];
    if (res) t += res[base + c];
    v[i] = t; s += t;
  }
  red[tid] = s; __syncthreads();
  for (int o = 128; o > 0; o >>= 1) { if (tid < o) red[tid] += red[tid + o]; __syncthreads(); }
  float mean = red[0] * (1.0f / HD);
  __syncthreads();
  float s2 = 0.0f;
#pragma unroll
  for (int i = 0; i < 8; ++i) { float d = v[i] - mean; s2 += d * d; }
  red[tid] = s2; __syncthreads();
  for (int o = 128; o > 0; o >>= 1) { if (tid < o) red[tid] += red[tid + o]; __syncthreads(); }
  float rstd = rsqrtf(red[0] * (1.0f / HD) + 1e-5f);
#pragma unroll
  for (int i = 0; i < 8; ++i) {
    int c = tid + i * 256;
    float y = (v[i] - mean) * rstd * gamma[c] + beta[c];
    out[base + c]   = y;
    outbf[base + c] = f2bf(y);
  }
}

// =====================================================================
extern "C" void kernel_launch(void* const* d_in, const int* in_sizes, int n_in,
                              void* d_out, int out_size, void* d_ws, size_t ws_size,
                              hipStream_t stream)
{
  const int*   ids   = (const int*)  d_in[0];
  const float* tok   = (const float*)d_in[1];
  const float* pos   = (const float*)d_in[2];
  const float* Wq    = (const float*)d_in[3];
  const float* bq    = (const float*)d_in[4];
  const float* Wk    = (const float*)d_in[5];
  const float* bk    = (const float*)d_in[6];
  const float* Wv    = (const float*)d_in[7];
  const float* bv    = (const float*)d_in[8];
  const float* Wo    = (const float*)d_in[9];
  const float* bo    = (const float*)d_in[10];
  const float* g1    = (const float*)d_in[11];
  const float* be1   = (const float*)d_in[12];
  const float* W1    = (const float*)d_in[13];
  const float* b1    = (const float*)d_in[14];
  const float* W2    = (const float*)d_in[15];
  const float* b2    = (const float*)d_in[16];
  const float* g2    = (const float*)d_in[17];
  const float* be2   = (const float*)d_in[18];
  const float* lnf_g = (const float*)d_in[19];
  const float* lnf_b = (const float*)d_in[20];
  const float* Whead = (const float*)d_in[21];
  float* out = (float*)d_out;

  // ---- workspace layout: f32 region then 16B-aligned bf16 region (~101MB) --
  float* ws   = (float*)d_ws;
  float* h    = ws;                                    // [S, HD]   f32
  float* tmp  = h    + (size_t)S_LEN * HD;             // [S, HD]   f32
  float* big  = tmp  + (size_t)S_LEN * HD;             // [S, FFD]  f32 (q proj)
  float* kb   = big  + (size_t)S_LEN * FFD;            // [S, GD]   f32
  float* vb   = kb   + (size_t)S_LEN * GD;             // [S, GD]   f32
  float* sc   = vb   + (size_t)S_LEN * GD;             // [G, S, S] f32
  float* wof  = sc   + (size_t)NGROUP * S_LEN * S_LEN; // [GD, HD]  f32

  unsigned short* bfbase = (unsigned short*)(wof + (size_t)GD * HD);
  unsigned short* hbf    = bfbase;                         // [S, HD]
  unsigned short* tmpbf  = hbf   + (size_t)S_LEN * HD;     // [S, HD]
  unsigned short* bigbf  = tmpbf + (size_t)S_LEN * HD;     // [S, FFD]
  unsigned short* qsbf   = bigbf + (size_t)S_LEN * FFD;    // [S, GD]
  unsigned short* cxbf   = qsbf  + (size_t)S_LEN * GD;     // [S, GD]
  unsigned short* scbf   = cxbf  + (size_t)S_LEN * GD;     // [G, S, S]

  auto gemm = [&](const unsigned short* A, int lda, const float* Bm, int ldb,
                  int btr, const float* bias, void* Y, int ldc,
                  int M, int N, int K, float scale, int relu, int outbf) {
    dim3 grid(M / 128, N / 64);   // M fastest: B column-tile reuse in L2
    wmma_gemm_kernel<<<grid, 256, 0, stream>>>(A, lda, Bm, ldb, btr, bias,
                                               Y, ldc, M, N, K, scale, relu, outbf);
  };

  // ---- embedding ----
  embed_kernel<<<dim3(S_LEN, HD / 256), 256, 0, stream>>>(ids, tok, pos, h, hbf);

  for (int l = 0; l < NLAYER; ++l) {
    const float* Wq_l = Wq + (size_t)l * HD * HD;   const float* bq_l = bq + (size_t)l * HD;
    const float* Wk_l = Wk + (size_t)l * HD * GD;   const float* bk_l = bk + (size_t)l * GD;
    const float* Wv_l = Wv + (size_t)l * HD * GD;   const float* bv_l = bv + (size_t)l * GD;
    const float* Wo_l = Wo + (size_t)l * HD * HD;   const float* bo_l = bo + (size_t)l * HD;
    const float* g1_l = g1 + (size_t)l * HD;        const float* be1_l = be1 + (size_t)l * HD;
    const float* W1_l = W1 + (size_t)l * HD * FFD;  const float* b1_l = b1 + (size_t)l * FFD;
    const float* W2_l = W2 + (size_t)l * FFD * HD;  const float* b2_l = b2 + (size_t)l * HD;
    const float* g2_l = g2 + (size_t)l * HD;        const float* be2_l = be2 + (size_t)l * HD;

    // QKV projections (A = bf16 h)
    gemm(hbf, HD, Wq_l, HD, 0, bq_l, big, HD, S_LEN, HD, HD, 1.0f, 0, 0);
    gemm(hbf, HD, Wk_l, GD, 0, bk_l, kb,  GD, S_LEN, GD, HD, 1.0f, 0, 0);
    gemm(hbf, HD, Wv_l, GD, 0, bv_l, vb,  GD, S_LEN, GD, HD, 1.0f, 0, 0);

    // reduce q over subheads -> bf16 qsum
    qreduce_kernel<<<S_LEN, 256, 0, stream>>>(big, qsbf);

    // scores[g] = qsum_g (SxD) @ k_g^T (DxS) / sqrt(D);  B stored N-major
    for (int g = 0; g < NGROUP; ++g)
      gemm(qsbf + g * HEADD, GD, kb + g * HEADD, GD, /*btrans=*/1, nullptr,
           sc + (size_t)g * S_LEN * S_LEN, S_LEN, S_LEN, S_LEN, HEADD, 0.125f, 0, 0);

    // softmax over t; emits bf16 probs for the ctx GEMM
    softmax_kernel<<<NGROUP * S_LEN, 256, 0, stream>>>(sc, scbf);

    // ctx[g] = attn_g (SxS) @ v_g (SxD) -> bf16 ctx
    for (int g = 0; g < NGROUP; ++g)
      gemm(scbf + (size_t)g * S_LEN * S_LEN, S_LEN, vb + g * HEADD, GD, 0, nullptr,
           cxbf + g * HEADD, GD, S_LEN, HEADD, S_LEN, 1.0f, 0, 1);

    // broadcast(ctx) @ Wo == ctx @ fold_j(Wo)
    wofold_kernel<<<dim3(HD / 256, GD), 256, 0, stream>>>(Wo_l, wof);
    gemm(cxbf, GD, wof, HD, 0, bo_l, tmp, HD, S_LEN, HD, GD, 1.0f, 0, 0);

    // h = LN(h + attn_out)  (f32 + bf16)
    add_ln_kernel<<<S_LEN, 256, 0, stream>>>(h, tmp, g1_l, be1_l, h, hbf);

    // FFN: relu GEMM emits bf16 directly (only consumed as A of FFN2)
    gemm(hbf,   HD,  W1_l, FFD, 0, b1_l, bigbf, FFD, S_LEN, FFD, HD,  1.0f, 1, 1);
    gemm(bigbf, FFD, W2_l, HD,  0, b2_l, tmp,   HD,  S_LEN, HD,  FFD, 1.0f, 0, 0);

    // h = LN(h + ffn)
    add_ln_kernel<<<S_LEN, 256, 0, stream>>>(h, tmp, g2_l, be2_l, h, hbf);
  }

  // final LN (no residual) -> tmp/tmpbf, then lm head into d_out (f32)
  add_ln_kernel<<<S_LEN, 256, 0, stream>>>(h, nullptr, lnf_g, lnf_b, tmp, tmpbf);
  gemm(tmpbf, HD, Whead, VOCAB, 0, nullptr, out, VOCAB, S_LEN, VOCAB, HD, 1.0f, 0, 0);
}
